// GumbelQuantizer_20684562498301
// MI455X (gfx1250) — matrix operand order repaired and checked by reading, not compiled, not run on hardware
//
#include <hip/hip_runtime.h>
#include <stdint.h>

// ---------------------------------------------------------------------------
// GumbelQuantizer fused kernel for gfx1250 (MI455X), wave32 + WMMA bf16.
//  h:[B,L,D] gumbel_u:[B,N,L] conv_w:[N,D] conv_b:[N] codebook:[N,CD]
//  out: sampled [B,L,CD] f32, then codes [B,L] i32 (concatenated in d_out)
//
//  Prep kernels convert conv_w / codebook to bf16 in d_ws, pre-swizzled into
//  WMMA A-/B-operand lane order: GEMM loops are b128-load(+imm-offset) + wmma.
//  Softmax is carried in base-2 scale (s2 = s * log2e / tau) so exp is a
//  single v_exp_f32 and logs are single v_log_f32 (args never denormal).
// ---------------------------------------------------------------------------

typedef __attribute__((ext_vector_type(16))) __bf16 v16bf;
typedef __attribute__((ext_vector_type(8)))  float  v8f;

#define B_   4
#define L_   2048
#define D_   1024
#define N_   8192
#define CD_  1024
#define EPS_ 1e-8f
#define LN2_    0.6931471805599453f
#define SCALE2_ 0.14426950408889634f   /* log2(e) / TAU */

// LDS row strides (elements), padded by 16 b16 to stagger banks
#define HPAD 1040     // 16 x 1040 x 2B = 33,280 B   (h tile, bf16)
#define SPAD 8208     // 16 x 8208 x 2B = 262,656 B  (score/p tile, bf16)
#define SM_BYTES (16*HPAD*2 + 16*SPAD*2 + 8*16*4*3 + 16*4*2)   // ~297.6 KB < 320 KB

#define WS_NEEDED ((size_t)(N_*(size_t)D_ + N_*(size_t)CD_) * 2)   // 32 MB

union BV { v16bf v; uint16_t u[16]; uint4 q[2]; };

static __device__ __forceinline__ uint16_t f2bf(float f) {
  union { __bf16 b; uint16_t u; } x; x.b = (__bf16)f;
  return x.u;
}
static __device__ __forceinline__ float bf2f(uint16_t h) {
  union { uint32_t u; float f; } x; x.u = ((uint32_t)h) << 16;
  return x.f;
}

// ======================= prep: conv_w -> A-operand bf16 ====================
// group g = ((ntile*32 + c)*32 + lane); element e of lane:
//   n = ntile*16 + (lane&15)
//   k = c*32 + 8*(lane>>4) + (e<8 ? e : e+8)
__global__ __launch_bounds__(256)
void prep_w_kernel(const float* __restrict__ w, uint16_t* __restrict__ wsw) {
  const int g    = blockIdx.x * 256 + threadIdx.x;   // 512*32*32 = 524288 groups
  const int lane = g & 31;
  const int c    = (g >> 5) & 31;
  const int nt   = g >> 10;
  const int n    = (nt << 4) + (lane & 15);
  const int k    = (c << 5) + ((lane >> 4) << 3);
  const float* src = w + (size_t)n * D_ + k;
  float4 a0 = *(const float4*)(src +  0);
  float4 a1 = *(const float4*)(src +  4);
  float4 a2 = *(const float4*)(src + 16);
  float4 a3 = *(const float4*)(src + 20);
  union { uint4 q[2]; uint16_t u[16]; } t;
  t.u[0]=f2bf(a0.x);  t.u[1]=f2bf(a0.y);  t.u[2]=f2bf(a0.z);  t.u[3]=f2bf(a0.w);
  t.u[4]=f2bf(a1.x);  t.u[5]=f2bf(a1.y);  t.u[6]=f2bf(a1.z);  t.u[7]=f2bf(a1.w);
  t.u[8]=f2bf(a2.x);  t.u[9]=f2bf(a2.y);  t.u[10]=f2bf(a2.z); t.u[11]=f2bf(a2.w);
  t.u[12]=f2bf(a3.x); t.u[13]=f2bf(a3.y); t.u[14]=f2bf(a3.z); t.u[15]=f2bf(a3.w);
  uint16_t* dst = wsw + (size_t)g * 16;
  *(uint4*)dst = t.q[0]; *(uint4*)(dst + 8) = t.q[1];
}

// ===================== prep: codebook -> B-operand bf16 ====================
// group g = ((nblk*64 + dtile)*32 + lane); element e of lane:
//   n = nblk*32 + 16*(lane>>4) + e
//   d = dtile*16 + (lane&15)
__global__ __launch_bounds__(256)
void prep_cb_kernel(const float* __restrict__ cb, uint16_t* __restrict__ cbsw) {
  const int g    = blockIdx.x * 256 + threadIdx.x;   // 256*64*32 = 524288 groups
  const int lane = g & 31;
  const int dt   = (g >> 5) & 63;
  const int nb   = g >> 11;
  const int d    = (dt << 4) + (lane & 15);
  const int nrow = (nb << 5) + ((lane >> 4) << 4);
  const float* src = cb + (size_t)nrow * CD_ + d;
  union { uint4 q[2]; uint16_t u[16]; } t;
#pragma unroll
  for (int e = 0; e < 16; ++e) t.u[e] = f2bf(src[(size_t)e * CD_]);
  uint16_t* dst = cbsw + (size_t)g * 16;
  *(uint4*)dst = t.q[0]; *(uint4*)(dst + 8) = t.q[1];
}

// ============================ main fused kernel ============================
template <bool PRE>
__global__ __launch_bounds__(256)
void gumbel_quantizer_fused(const float* __restrict__ h,
                            const float* __restrict__ gu,
                            const float* __restrict__ w,
                            const float* __restrict__ bias,
                            const float* __restrict__ cb,
                            const uint16_t* __restrict__ wsw,
                            const uint16_t* __restrict__ cbsw,
                            float* __restrict__ out_sampled,
                            int*   __restrict__ out_codes)
{
  extern __shared__ char smem[];
  uint16_t* hbuf   = (uint16_t*)smem;                               // [16][HPAD]
  uint16_t* sbuf   = (uint16_t*)(smem + 16*HPAD*2);                 // [16][SPAD]
  float*    redmax = (float*)(smem + 16*HPAD*2 + 16*SPAD*2);        // [8][16]
  int*      redarg = (int*)  (redmax + 8*16);                       // [8][16]
  float*    redsum = (float*)(redarg + 8*16);                       // [8][16]
  float*    fin_m  = redsum + 8*16;                                 // [16]
  float*    fin_iv = fin_m + 16;                                    // [16]

  const int tid  = threadIdx.x;
  const int wave = tid >> 5;
  const int lane = tid & 31;
  const int lh   = lane >> 4;      // lane half (0/1)
  const int ll   = lane & 15;      // lane low 4 bits
  const int b    = blockIdx.x >> 7;          // L_/16 == 128 tiles
  const int l0   = (blockIdx.x & 127) << 4;

  // ---- stage h tile [16 x 1024] fp32 -> bf16 LDS -------------------------
  {
    const int row = tid >> 4;               // 16 rows, 16 threads each
    const int c0  = (tid & 15) * 64;        // 64 contiguous cols per thread
    const float* src = h + ((size_t)b * L_ + (l0 + row)) * D_ + c0;
    uint16_t* dst = hbuf + row * HPAD + c0;
#pragma unroll
    for (int i = 0; i < 64; i += 4) {
      float4 v = *(const float4*)(src + i);
      dst[i+0] = f2bf(v.x); dst[i+1] = f2bf(v.y);
      dst[i+2] = f2bf(v.z); dst[i+3] = f2bf(v.w);
    }
  }
  __syncthreads();

  // ---- hoist GEMM-1 B operand (h tile) into registers: 32 K-chunks -------
  // B layout (32x16 bf16): lane j -> col l = j&15, e<8: K = 32c + 8*(j>>4) + e ; e>=8: +16
  v16bf Bh[32];
#pragma unroll
  for (int c = 0; c < 32; ++c) {
    const uint16_t* p = hbuf + ll * HPAD + c * 32 + lh * 8;
    BV t; t.q[0] = *(const uint4*)p; t.q[1] = *(const uint4*)(p + 16);
    Bh[c] = t.v;
  }

  // ======================= Phase 1: logits GEMM + gumbel ==================
  // Per wave: 64 n-tiles (n0 = (8i+wave)*16). D-tile: M=n, N=l (coalesced l).
  // Scores are kept in base-2 scale: s2 = (logit + g) * (log2e / TAU).
  float runmax = -__builtin_inff();
  int   runarg = 0;

  for (int i = 0; i < 64; ++i) {
    const int n0 = ((i << 3) + wave) << 4;
    v8f acc = {};
    // single base pointer; k-chunk stride is 512 elems = 1024 B (imm offsets)
    const uint16_t* abase = PRE ? (wsw + ((size_t)n0 * 64 + lane) * 16) : nullptr;
#pragma unroll
    for (int c = 0; c < 32; ++c) {
      BV A;
      if constexpr (PRE) {
        const uint16_t* ap = abase + c * 512;
        A.q[0] = *(const uint4*)ap; A.q[1] = *(const uint4*)(ap + 8);
      } else {
        // A layout (16x32 bf16): lane j -> row n0+(j&15),
        //   e<8 : K = 32c + 8*lh + e ; e>=8 : K = 32c + 16 + 8*lh + (e-8)
        const float* ap = w + (size_t)(n0 + ll) * D_ + c * 32 + lh * 8;
        float4 a0 = *(const float4*)(ap +  0);
        float4 a1 = *(const float4*)(ap +  4);
        float4 a2 = *(const float4*)(ap + 16);
        float4 a3 = *(const float4*)(ap + 20);
        A.u[0]=f2bf(a0.x);  A.u[1]=f2bf(a0.y);  A.u[2]=f2bf(a0.z);  A.u[3]=f2bf(a0.w);
        A.u[4]=f2bf(a1.x);  A.u[5]=f2bf(a1.y);  A.u[6]=f2bf(a1.z);  A.u[7]=f2bf(a1.w);
        A.u[8]=f2bf(a2.x);  A.u[9]=f2bf(a2.y);  A.u[10]=f2bf(a2.z); A.u[11]=f2bf(a2.w);
        A.u[12]=f2bf(a3.x); A.u[13]=f2bf(a3.y); A.u[14]=f2bf(a3.z); A.u[15]=f2bf(a3.w);
      }
      acc = __builtin_amdgcn_wmma_f32_16x16x32_bf16(
              false, A.v, false, Bh[c], (short)0, acc, false, false);
    }
    // epilogue: + bias + gumbel, to base-2 scale, store bf16, track max/argmax
    // D layout: lane j, reg r -> n = n0 + 8*(j>>4) + r, l = l0 + (j&15)
    const float* gup = gu + ((size_t)b * N_ + n0 + 8 * lh) * L_ + (l0 + ll);
    const float* bip = bias + n0 + 8 * lh;
    union { uint4 q; uint16_t u[8]; } st;
#pragma unroll
    for (int r = 0; r < 8; ++r) {
      const int n = n0 + 8 * lh + r;
      float logit = acc[r] + bip[r];
      float u = gup[(size_t)r * L_];
      // gumbel: g = -ln(-ln(u+eps)+eps); args never denormal -> raw v_log_f32
      float ln_u  = __log2f(u + EPS_) * LN2_;
      float inner = EPS_ - ln_u;
      float g     = -(__log2f(inner) * LN2_);
      float s2    = (logit + g) * SCALE2_;
      st.u[r] = f2bf(s2);
      if (s2 > runmax || (s2 == runmax && n < runarg)) { runmax = s2; runarg = n; }
    }
    *(uint4*)(sbuf + ll * SPAD + n0 + 8 * lh) = st.q;   // 8 contiguous bf16
  }
  // merge lane halves (same l, disjoint n)
  {
    float om = __shfl_xor(runmax, 16, 32);
    int   oa = __shfl_xor(runarg, 16, 32);
    if (om > runmax || (om == runmax && oa < runarg)) { runmax = om; runarg = oa; }
    if (lane < 16) { redmax[wave * 16 + ll] = runmax; redarg[wave * 16 + ll] = runarg; }
  }
  __syncthreads();

  if (tid < 16) {            // cross-wave max/argmax; emit codes
    float m = -__builtin_inff(); int a = 0;
    for (int wv = 0; wv < 8; ++wv) {
      float x = redmax[wv * 16 + tid]; int ai = redarg[wv * 16 + tid];
      if (x > m || (x == m && ai < a)) { m = x; a = ai; }
    }
    fin_m[tid] = m;
    out_codes[(size_t)b * L_ + l0 + tid] = a;
  }
  __syncthreads();

  // ============ Phase 2a: p = exp2(s2 - m2) in-place, row sums ============
  {
    const float m_l = fin_m[ll];
    const int nbase = wave * 1024 + lh * 512;   // disjoint n slice per (wave,half)
    float psum = 0.f;
    for (int k = 0; k < 512; k += 8) {
      uint16_t* p0 = sbuf + ll * SPAD + nbase + k;
      union { uint4 q; uint16_t u[8]; } t; t.q = *(const uint4*)p0;
      float ls = 0.f;
#pragma unroll
      for (int e = 0; e < 8; ++e) {
        float p = __builtin_amdgcn_exp2f(bf2f(t.u[e]) - m_l);  // single v_exp_f32
        t.u[e] = f2bf(p);
        ls += p;
      }
      *(uint4*)p0 = t.q;
      psum += ls;
    }
    psum += __shfl_xor(psum, 16, 32);
    if (lane < 16) redsum[wave * 16 + ll] = psum;
  }
  __syncthreads();
  if (tid < 16) {
    float s = 0.f;
    for (int wv = 0; wv < 8; ++wv) s += redsum[wv * 16 + tid];
    fin_iv[tid] = 1.0f / s;
  }
  __syncthreads();

  // ====== Phase 2b: sampled = (p · codebook) * (1/sum), CD split/wave =====
  const int d0 = wave * 128;   // each wave owns 128 CD columns = 8 N-tiles
  v8f zero = {};
  v8f acc2[8];
#pragma unroll
  for (int t = 0; t < 8; ++t) acc2[t] = zero;

  for (int n0 = 0; n0 < N_; n0 += 32) {
    // A (p) layout: lane j -> row l = j&15 ; K = n0 + 16*lh + e
    const uint16_t* pp = sbuf + ll * SPAD + n0 + 16 * lh;
    BV A; A.q[0] = *(const uint4*)pp; A.q[1] = *(const uint4*)(pp + 8);
    // single base pointer; d-tile stride is 512 elems = 1024 B (imm offsets)
    const uint16_t* bbase =
        PRE ? (cbsw + ((size_t)n0 * 64 + wave * 256 + lane) * 16) : nullptr;
#pragma unroll
    for (int t = 0; t < 8; ++t) {
      BV Bt;
      if constexpr (PRE) {
        const uint16_t* bp = bbase + t * 512;
        Bt.q[0] = *(const uint4*)bp; Bt.q[1] = *(const uint4*)(bp + 8);
      } else {
        const int dc = d0 + t * 16 + ll;
        // B (codebook) layout: lane j -> col dc, K = n0 + 16*lh + e
        const float* cp = cb + (size_t)(n0 + 16 * lh) * CD_ + dc;
#pragma unroll
        for (int e = 0; e < 16; ++e) Bt.u[e] = f2bf(cp[(size_t)e * CD_]);
      }
      acc2[t] = __builtin_amdgcn_wmma_f32_16x16x32_bf16(
                  false, A.v, false, Bt.v, (short)0, acc2[t], false, false);
    }
  }

  // epilogue: normalize & store. D layout: lane j, reg r -> l = r+8*(j>>4)
#pragma unroll
  for (int t = 0; t < 8; ++t) {
    const int dc = d0 + t * 16 + ll;
#pragma unroll
    for (int r = 0; r < 8; ++r) {
      const int l = r + 8 * lh;
      out_sampled[((size_t)b * L_ + (l0 + l)) * CD_ + dc] = acc2[t][r] * fin_iv[l];
    }
  }
}

extern "C" void kernel_launch(void* const* d_in, const int* in_sizes, int n_in,
                              void* d_out, int out_size, void* d_ws, size_t ws_size,
                              hipStream_t stream) {
  (void)in_sizes; (void)n_in; (void)out_size;
  const float* h    = (const float*)d_in[0];
  const float* gu   = (const float*)d_in[1];
  const float* cw   = (const float*)d_in[2];
  const float* cbi  = (const float*)d_in[3];
  const float* code = (const float*)d_in[4];

  float* sampled = (float*)d_out;
  int*   codes   = (int*)((float*)d_out + (size_t)B_ * L_ * CD_);

  uint16_t* wsw  = (uint16_t*)d_ws;
  uint16_t* cbsw = wsw + (size_t)N_ * D_;

  (void)hipFuncSetAttribute((const void*)gumbel_quantizer_fused<true>,
                            hipFuncAttributeMaxDynamicSharedMemorySize, SM_BYTES);
  (void)hipFuncSetAttribute((const void*)gumbel_quantizer_fused<false>,
                            hipFuncAttributeMaxDynamicSharedMemorySize, SM_BYTES);

  dim3 grid(B_ * (L_ / 16));   // 512 workgroups
  dim3 block(256);             // 8 wave32

  if (ws_size >= WS_NEEDED) {
    prep_w_kernel <<<2048, 256, 0, stream>>>(cw,  wsw);
    prep_cb_kernel<<<2048, 256, 0, stream>>>(cbi, cbsw);
    gumbel_quantizer_fused<true><<<grid, block, SM_BYTES, stream>>>(
        h, gu, cw, cbi, code, wsw, cbsw, sampled, codes);
  } else {
    gumbel_quantizer_fused<false><<<grid, block, SM_BYTES, stream>>>(
        h, gu, cw, cbi, code, wsw, cbsw, sampled, codes);
  }
}
